// ASI_block_att_48885317763141
// MI455X (gfx1250) — compile-verified, matrix operand order checked by reading, not run on hardware
//
#include <hip/hip_runtime.h>
#include <math.h>

typedef __attribute__((ext_vector_type(16))) _Float16 v16h;
typedef __attribute__((ext_vector_type(8)))  _Float16 v8h;
typedef __attribute__((ext_vector_type(8)))  float    v8f;

// LDS (addrspace 3) qualified element types: keeps every LDS access on the
// ds_* path with 32-bit addressing -- no generic-pointer null-check selects.
#define LDS_AS __attribute__((address_space(3)))
typedef LDS_AS _Float16 lh;
typedef LDS_AS float    lf;
typedef LDS_AS v8h      lv8h;

#define NB     16
#define NID    170
#define NHI    128
#define DF     96
#define EPSV   1e-5f

// LDS layout (byte offsets)
#define OFF_XF32   0          // 128*96 f32 = 49152
#define OFF_Y1     49152      // 128*96 f16 = 24576
#define OFF_Y2     73728
#define OFF_QS     98304
#define OFF_KS     122880
#define OFF_VT     147456     // 96*128 f16 (transposed V)
#define OFF_PS     172032     // 128*128 f16 = 32768
#define OFF_WS     204800     // 96*96 f16 = 18432
#define OFF_BS     223232     // 96 f32
#define OFF_LNW    223616
#define OFF_LNB    224000
#define SMEM_BYTES 224384

struct AttP { const float *wq,*bq,*wk,*bk,*wv,*bv,*w1,*b1; };
struct KParams {
  const float* x;
  AttP a[4];                      // t1, t2, i1, i2
  const float* tlnw[2]; const float* tlnb[2];
  const float* lw[2];   const float* lb[2];
  float* out;
};

// ---- WMMA fragment load -----------------------------------------------------
// A operand (16x32, M over lane%16) and B operand (32x16, N over lane%16) share
// the same per-lane layout; source is a row-major [m_or_n][k] f16 panel in LDS.
// Lanes 0-15 cover K {0..7,16..23}, lanes 16-31 cover K {8..15,24..31}.
template<int LD>
__device__ inline v16h load_frag(const lh* base) {
  const int lane = threadIdx.x & 31;
  const lh* p = base + (lane & 15) * LD + ((lane & 16) ? 8 : 0);
  v16h f;
#pragma unroll
  for (int h = 0; h < 8; ++h)  f[h]     = p[h];
#pragma unroll
  for (int h = 0; h < 8; ++h)  f[h + 8] = p[h + 16];
  return f;
}

// Preload the wave-invariant A row-panel fragments once per GEMM.
template<int K, int LD>
__device__ inline void load_afrags(const lh* A, v16h* out) {
#pragma unroll
  for (int k = 0; k < K / 32; ++k) out[k] = load_frag<LD>(A + k * 32);
}

// GEMM with preloaded A fragments; B streamed from LDS per tile.
template<int K, int LDB>
__device__ inline v8f gemm_pre(const v16h* a, const lh* Bt, v8f acc) {
#pragma unroll
  for (int k = 0; k < K / 32; ++k) {
    v16h b = load_frag<LDB>(Bt + k * 32);
    acc = __builtin_amdgcn_wmma_f32_16x16x32_f16(false, a[k], false, b,
                                                 (short)0, acc, false, false);
  }
  return acc;
}

__device__ inline float rsum16(float v) {
#pragma unroll
  for (int m = 1; m < 16; m <<= 1) v += __shfl_xor(v, m, 32);
  return v;
}
__device__ inline float rmax16(float v) {
#pragma unroll
  for (int m = 1; m < 16; m <<= 1) v = fmaxf(v, __shfl_xor(v, m, 32));
  return v;
}

// ---- weight panel -> LDS (f16) ---------------------------------------------
__device__ inline void load_w(const float* wsrc, const float* bsrc,
                              lh* Ws, lf* Bs, int tid) {
  for (int i = tid; i < DF * DF; i += 256) Ws[i] = (_Float16)wsrc[i];
  for (int i = tid; i < DF;      i += 256) Bs[i] = bsrc[i];
}

// ---- linear: Out(128x96) = A(128x96) @ W^T + b ------------------------------
template<bool TRANSP, int LDO>
__device__ inline void linear96(const lh* Asrc, const lh* Ws,
                                const lf* Bs, lh* Out,
                                int wave, int lane) {
  const int ncol = lane & 15;
  const int mo   = wave * 16 + ((lane & 16) ? 8 : 0);
  v16h af[DF / 32];
  load_afrags<DF, DF>(Asrc + wave * 16 * DF, af);
#pragma unroll
  for (int nt = 0; nt < 6; ++nt) {
    v8f acc;
    float bv = Bs[nt * 16 + ncol];
#pragma unroll
    for (int r = 0; r < 8; ++r) acc[r] = bv;
    acc = gemm_pre<DF, DF>(af, Ws + nt * 16 * DF, acc);
    const int n = nt * 16 + ncol;
    if (TRANSP) {
      // rows are contiguous in destination: one 16-byte packed store
      v8h pk;
#pragma unroll
      for (int r = 0; r < 8; ++r) pk[r] = (_Float16)acc[r];
      *(lv8h*)(Out + n * LDO + mo) = pk;
    } else {
#pragma unroll
      for (int r = 0; r < 8; ++r) Out[(mo + r) * LDO + n] = (_Float16)acc[r];
    }
  }
}

__device__ void qkv_stage(const lh* Qsrc, const lh* KVsrc,
                          lh* Qs, lh* Ks, lh* Vt,
                          lh* Ws, lf* Bs, const AttP& ap,
                          int wave, int tid) {
  const int lane = tid & 31;
  __syncthreads();
  load_w(ap.wq, ap.bq, Ws, Bs, tid); __syncthreads();
  linear96<false, DF>(Qsrc, Ws, Bs, Qs, wave, lane); __syncthreads();
  load_w(ap.wk, ap.bk, Ws, Bs, tid); __syncthreads();
  linear96<false, DF>(KVsrc, Ws, Bs, Ks, wave, lane); __syncthreads();
  load_w(ap.wv, ap.bv, Ws, Bs, tid); __syncthreads();
  linear96<true, NHI>(KVsrc, Ws, Bs, Vt, wave, lane); __syncthreads();
}

// ---- attention core: O = softmax(Q K^T / 4) V, head-combined ----------------
__device__ void att_core(const lh* Qs, const lh* Ks,
                         const lh* Vt, lh* Ps,
                         int wave, int lane, float sh, float bh,
                         float O[6][8]) {
  float S[8][8];
  {
    v16h qf[DF / 32];
    load_afrags<DF, DF>(Qs + wave * 16 * DF, qf);   // this wave's Q rows, reused 8x
#pragma unroll
    for (int jt = 0; jt < 8; ++jt) {
      v8f acc;
#pragma unroll
      for (int r = 0; r < 8; ++r) acc[r] = 0.f;
      acc = gemm_pre<DF, DF>(qf, Ks + jt * 16 * DF, acc);
#pragma unroll
      for (int r = 0; r < 8; ++r) S[jt][r] = acc[r] * 0.25f; // kd = sqrt(128/8)
    }
  }
  // softmax per row (row elems: 16 lanes x 8 j-tiles)
#pragma unroll
  for (int r = 0; r < 8; ++r) {
    float mx = S[0][r];
#pragma unroll
    for (int jt = 1; jt < 8; ++jt) mx = fmaxf(mx, S[jt][r]);
    mx = rmax16(mx);
    float sum = 0.f;
#pragma unroll
    for (int jt = 0; jt < 8; ++jt) { S[jt][r] = __expf(S[jt][r] - mx); sum += S[jt][r]; }
    sum = rsum16(sum);
    float inv = 1.0f / sum;
    int m = wave * 16 + r + ((lane & 16) ? 8 : 0);
#pragma unroll
    for (int jt = 0; jt < 8; ++jt)
      Ps[m * NHI + jt * 16 + (lane & 15)] = (_Float16)(S[jt][r] * inv);
  }
  // O = P V  (A rows belong to this wave; Vt barrier-covered by qkv_stage)
  {
    v16h pf[NHI / 32];
    load_afrags<NHI, NHI>(Ps + wave * 16 * NHI, pf); // this wave's P rows, reused 6x
#pragma unroll
    for (int nt = 0; nt < 6; ++nt) {
      v8f acc;
#pragma unroll
      for (int r = 0; r < 8; ++r) acc[r] = 0.f;
      acc = gemm_pre<NHI, NHI>(pf, Vt + nt * 16 * NHI, acc);
#pragma unroll
      for (int r = 0; r < 8; ++r) O[nt][r] = acc[r] * sh + bh; // head combine
    }
  }
}

// ---- fused per-(b,id) kernel ------------------------------------------------
__global__ __launch_bounds__(256)
void asi_block_att_kernel(KParams P) {
  extern __shared__ char smem[];
  // Low 32 bits of a generic LDS address are the LDS byte offset (aperture
  // mapping); build 32-bit addrspace(3) pointers from it once.
  const unsigned lbase = (unsigned)(size_t)(void*)smem;
  lf* Xf32 = (lf*)(lbase + OFF_XF32);
  lh* Y0   = (lh*)(lbase + OFF_Y1);
  lh* Y1   = (lh*)(lbase + OFF_Y2);
  lh* Qs   = (lh*)(lbase + OFF_QS);
  lh* Ks   = (lh*)(lbase + OFF_KS);
  lh* Vt   = (lh*)(lbase + OFF_VT);
  lh* Ps   = (lh*)(lbase + OFF_PS);
  lh* Ws   = (lh*)(lbase + OFF_WS);
  lf* Bs   = (lf*)(lbase + OFF_BS);
  lf* LnW  = (lf*)(lbase + OFF_LNW);
  lf* LnB  = (lf*)(lbase + OFF_LNB);

  const int tid  = threadIdx.x;
  const int wave = tid >> 5;
  const int lane = tid & 31;
  const int b    = blockIdx.x / NID;
  const int nid  = blockIdx.x % NID;
  const size_t slab = (size_t)(b * NID + nid) * NHI;
  const float2* xpair = (const float2*)P.x;   // [slab+m][f] -> (ch0, ch1)

  // ===== time attention per channel =====
  for (int c = 0; c < 2; ++c) {
    const AttP& ap = P.a[c];
    lh* Yc = (c == 0) ? Y0 : Y1;
    for (int i = tid; i < NHI * DF; i += 256) {   // coalesced de-interleave
      int m = i / DF, f = i - m * DF;
      float2 v2 = xpair[(slab + m) * DF + f];
      float v = (c == 0) ? v2.x : v2.y;
      Xf32[i] = v;
      Yc[i] = (_Float16)v;
    }
    for (int i = tid; i < DF; i += 256) { LnW[i] = P.tlnw[c][i]; LnB[i] = P.tlnb[c][i]; }
    __syncthreads();
    qkv_stage(Yc, Yc, Qs, Ks, Vt, Ws, Bs, ap, wave, tid);
    float sh = 0.f;
#pragma unroll
    for (int i = 0; i < 8; ++i) sh += ap.w1[i];
    float bh = ap.b1[0];
    float O[6][8];
    att_core(Qs, Ks, Vt, Ps, wave, lane, sh, bh, O);
    // residual + per-row LayerNorm; result back into Yc (f16) for interaction
#pragma unroll
    for (int r = 0; r < 8; ++r) {
      int m = wave * 16 + r + ((lane & 16) ? 8 : 0);
      float val[6], s = 0.f;
#pragma unroll
      for (int t = 0; t < 6; ++t) {
        val[t] = Xf32[m * DF + t * 16 + (lane & 15)] + O[t][r];
        s += val[t];
      }
      s = rsum16(s);
      float mean = s * (1.f / DF);
      float q = 0.f;
#pragma unroll
      for (int t = 0; t < 6; ++t) { float d = val[t] - mean; q += d * d; }
      q = rsum16(q);
      float rstd = rsqrtf(q * (1.f / DF) + EPSV);
#pragma unroll
      for (int t = 0; t < 6; ++t) {
        int n = t * 16 + (lane & 15);
        Yc[m * DF + n] = (_Float16)((val[t] - mean) * rstd * LnW[n] + LnB[n]);
      }
    }
    __syncthreads();
  }

  // ===== interaction attention: x11 = att(Q=x2', KV=x1'), x22 = att(Q=x1', KV=x2') =====
  for (int c = 0; c < 2; ++c) {
    const AttP& ap = P.a[2 + c];
    const lh* Qsrc  = (c == 0) ? Y1 : Y0;
    const lh* KVsrc = (c == 0) ? Y0 : Y1;
    for (int i = tid; i < DF; i += 256) LnW[i] = P.lw[c][i];   // reuse for l_w
    qkv_stage(Qsrc, KVsrc, Qs, Ks, Vt, Ws, Bs, ap, wave, tid);
    float sh = 0.f;
#pragma unroll
    for (int i = 0; i < 8; ++i) sh += ap.w1[i];
    float bh = ap.b1[0];
    float O[6][8];
    att_core(Qs, Ks, Vt, Ps, wave, lane, sh, bh, O);
    // x += (x . l_w + l_b) broadcast, store pre-global-LN to out channel c
    float lb0 = P.lb[c][0];
#pragma unroll
    for (int r = 0; r < 8; ++r) {
      int m = wave * 16 + r + ((lane & 16) ? 8 : 0);
      float dotv = 0.f;
#pragma unroll
      for (int t = 0; t < 6; ++t) dotv += O[t][r] * LnW[t * 16 + (lane & 15)];
      dotv = rsum16(dotv) + lb0;
#pragma unroll
      for (int t = 0; t < 6; ++t) {
        int n = t * 16 + (lane & 15);
        P.out[((slab + m) * DF + n) * 2 + c] = O[t][r] + dotv;
      }
    }
    __syncthreads();
  }
}

// ---- global-LN pass 1: per (b, channel) mean / rstd -------------------------
__global__ __launch_bounds__(1024)
void asi_stats_kernel(const float* __restrict__ out, float* __restrict__ stats) {
  const int b = blockIdx.x, c = blockIdx.y;
  const size_t N = (size_t)NID * NHI * DF;       // 2,088,960
  const size_t base = (size_t)b * N;
  float s = 0.f, q = 0.f;
  for (size_t i = threadIdx.x; i < N; i += 1024) {
    float v = out[(base + i) * 2 + c];
    s += v; q += v * v;
  }
  __shared__ float shs[1024], shq[1024];
  shs[threadIdx.x] = s; shq[threadIdx.x] = q;
  __syncthreads();
  for (int st = 512; st > 0; st >>= 1) {
    if (threadIdx.x < st) {
      shs[threadIdx.x] += shs[threadIdx.x + st];
      shq[threadIdx.x] += shq[threadIdx.x + st];
    }
    __syncthreads();
  }
  if (threadIdx.x == 0) {
    float mean = shs[0] / (float)N;
    float var  = shq[0] / (float)N - mean * mean;
    stats[(b * 2 + c) * 2 + 0] = mean;
    stats[(b * 2 + c) * 2 + 1] = rsqrtf(var + EPSV);
  }
}

// ---- global-LN pass 2: normalize in place -----------------------------------
__global__ __launch_bounds__(256)
void asi_norm_kernel(float* __restrict__ out, const float* __restrict__ stats,
                     const float* __restrict__ w0, const float* __restrict__ b0,
                     const float* __restrict__ w1, const float* __restrict__ b1) {
  const size_t PER = (size_t)NID * NHI * DF;
  const size_t T   = (size_t)NB * PER;
  size_t stride = (size_t)gridDim.x * blockDim.x;
  for (size_t i = (size_t)blockIdx.x * blockDim.x + threadIdx.x; i < T; i += stride) {
    int b = (int)(i / PER);
    size_t r = i - (size_t)b * PER;
    float m0 = stats[(b * 2 + 0) * 2], r0 = stats[(b * 2 + 0) * 2 + 1];
    float m1 = stats[(b * 2 + 1) * 2], r1 = stats[(b * 2 + 1) * 2 + 1];
    float v0 = out[i * 2 + 0];
    float v1 = out[i * 2 + 1];
    out[i * 2 + 0] = (v0 - m0) * r0 * w0[r] + b0[r];
    out[i * 2 + 1] = (v1 - m1) * r1 * w1[r] + b1[r];
  }
}

extern "C" void kernel_launch(void* const* d_in, const int* in_sizes, int n_in,
                              void* d_out, int out_size, void* d_ws, size_t ws_size,
                              hipStream_t stream) {
  auto F = [&](int i) { return (const float*)d_in[i]; };
  // setup_inputs() flatten order: x, t1{wq,bq,wk,bk,wv,bv,w1,b1,ln_w,ln_b},
  // t2{...}, i1{wq..b1}, i2{wq..b1}, l1_w, l1_b, l2_w, l2_b, ln1_w, ln1_b, ln2_w, ln2_b
  KParams P;
  P.x = F(0);
  int idx = 1;
  for (int s = 0; s < 2; ++s) {  // t1, t2 (with ln)
    P.a[s] = { F(idx), F(idx+1), F(idx+2), F(idx+3), F(idx+4), F(idx+5), F(idx+6), F(idx+7) };
    P.tlnw[s] = F(idx + 8);
    P.tlnb[s] = F(idx + 9);
    idx += 10;
  }
  for (int s = 0; s < 2; ++s) {  // i1, i2 (no ln)
    P.a[2 + s] = { F(idx), F(idx+1), F(idx+2), F(idx+3), F(idx+4), F(idx+5), F(idx+6), F(idx+7) };
    idx += 8;
  }
  P.lw[0] = F(idx);   P.lb[0] = F(idx + 1);
  P.lw[1] = F(idx + 2); P.lb[1] = F(idx + 3);
  const float* ln1w = F(idx + 4);
  const float* ln1b = F(idx + 5);
  const float* ln2w = F(idx + 6);
  const float* ln2b = F(idx + 7);
  P.out = (float*)d_out;

  asi_block_att_kernel<<<NB * NID, 256, SMEM_BYTES, stream>>>(P);
  asi_stats_kernel<<<dim3(NB, 2), 1024, 0, stream>>>((const float*)d_out, (float*)d_ws);
  asi_norm_kernel<<<8192, 256, 0, stream>>>((float*)d_out, (const float*)d_ws,
                                            ln1w, ln1b, ln2w, ln2b);
}
